// SpatialPooler_14405320311077
// MI455X (gfx1250) — compile-verified, matrix operand order checked by reading, not run on hardware
//
#include <hip/hip_runtime.h>
#include <hip/hip_bf16.h>
#include <float.h>

// ---------------------------------------------------------------------------
// SpatialPooler step for MI455X (gfx1250, wave32, WMMA).
//
// Inputs (setup_inputs order):
//   d_in[0] x              (256 x 8192)  f32 (binary 0/1)
//   d_in[1] permanences    (4096 x 8192) f32
//   d_in[2] potential_mask (4096 x 8192) bool (1 byte each)
//   d_in[3] boost_weights  (4096 x 4096) f32 (rows sum to 1)
//   d_in[4] k              int scalar (40)
//   d_in[5] t_step         int scalar (2500)
// Outputs, concatenated flat in d_out (floats):
//   boosted   (256 x 4096)             @ 0
//   active_idx(256 x k) as float       @ 1048576
//   new_perm  (4096 x 8192)            @ 1048576 + 256*k
//
// Roofline: update pass streams ~300MB of perm/mask -> ~13us @ 23.3TB/s;
// both GEMMs total 34 GFLOP in bf16 WMMA, far under matrix-core ceiling.
// => memory-bound; fuse S = active.T @ x into the update, never spill S.
// ---------------------------------------------------------------------------

#define N_INP  8192
#define N_COLS 4096
#define BATCH  256
#define ONE_BF16 ((unsigned short)0x3F80)

typedef __attribute__((ext_vector_type(16))) __bf16          v16bf;
typedef __attribute__((ext_vector_type(8)))  float           v8f;
typedef __attribute__((ext_vector_type(8)))  unsigned short  u16x8;
typedef __attribute__((ext_vector_type(16))) unsigned short  u16x16;
typedef __attribute__((ext_vector_type(4)))  float           f32x4;

union FragU {
    u16x16 u;
    v16bf  v;
    struct Halves { u16x8 lo; u16x8 hi; } h;
};

__device__ __forceinline__ float gamma_of(int t) {
    float tf   = (float)t;
    float frac = fminf(fmaxf((tf - 1000.0f) / 5000.0f, 0.0f), 1.0f);
    float dec  = 0.2f + 0.8f * (1.0f - frac);
    return tf < 1000.0f ? 1.0f : dec;
}

// ISA 7.12.2: 16-bit A-matrix 16x32 per-lane K map (B assumed symmetric with
// lane = column). Lane half selects K base {0,8}; vector elems 0-7 -> K
// base+0..7, elems 8-15 -> K base+16..23.
__device__ __forceinline__ int kmap(int l, int j) {
    return ((l & 16) >> 1) + (j < 8 ? j : j + 8);
}

// --- zero per-launch accumulators (activeT bf16, count f32) ----------------
__global__ void k_init(unsigned short* __restrict__ activeT,
                       float* __restrict__ count) {
    int i = blockIdx.x * blockDim.x + threadIdx.x;
    if (i < N_COLS * BATCH) activeT[i] = 0;
    if (i < N_COLS)         count[i]   = 0.0f;
}

// --- boost[c] = 1 + gamma*(exp(BETA*(mu_c - duty)) - 1), mu = duty*rowsum --
__global__ __launch_bounds__(256) void k_boost(const float* __restrict__ bw,
                                               const int* __restrict__ kp,
                                               const int* __restrict__ tp,
                                               float* __restrict__ boost) {
    __shared__ float red[256];
    int c = blockIdx.x, t = threadIdx.x;
    float s = 0.0f;
    for (int j = t; j < N_COLS; j += 256) s += bw[(size_t)c * N_COLS + j];
    red[t] = s;
    __syncthreads();
    for (int st = 128; st > 0; st >>= 1) {
        if (t < st) red[t] += red[t + st];
        __syncthreads();
    }
    if (t == 0) {
        float duty = (float)(*kp) / (float)N_COLS;
        float mu   = duty * red[0];
        float g    = gamma_of(*tp);
        boost[c]   = 1.0f + g * (__expf(3.0f * (mu - duty)) - 1.0f);
    }
}

// --- connected bf16: perm >= 0.2 (mask redundant here: perm==0 off-pool) ---
__global__ void k_connected(const float* __restrict__ perm,
                            unsigned short* __restrict__ conn) {
    size_t i8 = ((size_t)blockIdx.x * blockDim.x + threadIdx.x) * 8;
    const f32x4* p = (const f32x4*)(perm + i8);
    f32x4 a = p[0], b = p[1];
    u16x8 o;
    o[0] = a[0] >= 0.2f ? ONE_BF16 : 0;  o[1] = a[1] >= 0.2f ? ONE_BF16 : 0;
    o[2] = a[2] >= 0.2f ? ONE_BF16 : 0;  o[3] = a[3] >= 0.2f ? ONE_BF16 : 0;
    o[4] = b[0] >= 0.2f ? ONE_BF16 : 0;  o[5] = b[1] >= 0.2f ? ONE_BF16 : 0;
    o[6] = b[2] >= 0.2f ? ONE_BF16 : 0;  o[7] = b[3] >= 0.2f ? ONE_BF16 : 0;
    *(u16x8*)(conn + i8) = o;
}

// --- x repacked into WMMA-B fragment order for GEMM2 (K=b, N=i) ------------
// tile = nt*8 + kt; element e = tile*512 + l*16 + j  <->  (b = kt*32+kmap,
// i = nt*16 + (l&15)).  One contiguous 32B load per lane at GEMM time.
__global__ void k_pack_x_g2(const float* __restrict__ x,
                            unsigned short* __restrict__ xw) {
    int e = blockIdx.x * blockDim.x + threadIdx.x;        // < 256*8192
    int tile = e >> 9, l = (e >> 4) & 31, j = e & 15;
    int kt = tile & 7, nt = tile >> 3;
    int b = kt * 32 + kmap(l, j);
    int i = nt * 16 + (l & 15);
    xw[e] = x[(size_t)b * N_INP + i] > 0.0f ? ONE_BF16 : 0;
}

// --- xT repacked into WMMA-B fragment order for GEMM1 (K=i, N=b) -----------
// tile = kt*16 + nt; element <-> (i = kt*32+kmap, b = nt*16 + (l&15)).
__global__ void k_pack_x_g1(const float* __restrict__ x,
                            unsigned short* __restrict__ xT) {
    int e = blockIdx.x * blockDim.x + threadIdx.x;        // < 256*8192
    int tile = e >> 9, l = (e >> 4) & 31, j = e & 15;
    int kt = tile >> 4, nt = tile & 15;
    int i = kt * 32 + kmap(l, j);
    int b = nt * 16 + (l & 15);
    xT[e] = x[(size_t)b * N_INP + i] > 0.0f ? ONE_BF16 : 0;
}

// --- GEMM1: overlapT = connected @ xT, boost applied, stored (b,c)-major ---
// One wave per 16x16 (c,b) tile; K = 8192 in steps of 32.
__global__ __launch_bounds__(256) void k_gemm1(
        const unsigned short* __restrict__ conn,
        const unsigned short* __restrict__ xT,
        const float* __restrict__ boost,
        float* __restrict__ boosted_out) {
    int w  = blockIdx.x * 8 + (threadIdx.x >> 5);
    int l  = threadIdx.x & 31;
    int mt = w >> 4, nt = w & 15;
    int c0 = mt * 16, b0 = nt * 16;
    const unsigned short* arow  = conn + (size_t)(c0 + (l & 15)) * N_INP;
    const unsigned short* bbase = xT + l * 16;
    int base = (l & 16) >> 1;
    v8f acc = {};
    for (int kt = 0; kt < N_INP / 32; ++kt) {
        FragU a, b;
        a.h.lo = *(const u16x8*)(arow + kt * 32 + base);
        a.h.hi = *(const u16x8*)(arow + kt * 32 + base + 16);
        b.u    = *(const u16x16*)(bbase + (size_t)(kt * 16 + nt) * 512);
        acc = __builtin_amdgcn_wmma_f32_16x16x32_bf16(
                  false, a.v, false, b.v, (short)0, acc, false, false);
    }
    int n = l & 15, hb = (l & 16) >> 1;
#pragma unroll
    for (int r = 0; r < 8; ++r) {
        int c = c0 + r + hb;
        boosted_out[(size_t)(b0 + n) * N_COLS + c] = acc[r] * boost[c];
    }
}

// --- exact top-k per batch row: iterative LDS argmax (lowest-index ties) ---
__global__ __launch_bounds__(256) void k_topk(
        const float* __restrict__ boosted, const int* __restrict__ kp,
        float* __restrict__ idx_out, unsigned short* __restrict__ activeT,
        float* __restrict__ count) {
    __shared__ float row[N_COLS];
    __shared__ float rv[256];
    __shared__ int   ri[256];
    int b = blockIdx.x, t = threadIdx.x;
    for (int i = t; i < N_COLS; i += 256)
        row[i] = boosted[(size_t)b * N_COLS + i];
    __syncthreads();
    int K = *kp;
    for (int sel = 0; sel < K; ++sel) {
        float bv = -FLT_MAX; int bi = 0x7FFFFFFF;
        for (int i = t; i < N_COLS; i += 256) {
            float v = row[i];
            if (v > bv || (v == bv && i < bi)) { bv = v; bi = i; }
        }
        rv[t] = bv; ri[t] = bi;
        __syncthreads();
        for (int s = 128; s > 0; s >>= 1) {
            if (t < s) {
                float ov = rv[t + s]; int oi = ri[t + s];
                if (ov > rv[t] || (ov == rv[t] && oi < ri[t])) {
                    rv[t] = ov; ri[t] = oi;
                }
            }
            __syncthreads();
        }
        int widx = ri[0];
        if (t == 0) {
            idx_out[b * K + sel] = (float)widx;
            activeT[(size_t)widx * BATCH + b] = ONE_BF16;   // A of GEMM2
            atomicAdd(&count[widx], 1.0f);                  // win counts
            row[widx] = -FLT_MAX;
        }
        __syncthreads();
    }
}

// --- GEMM2 fused update: S-tile = activeT @ x, then clipped Hebbian --------
// One wave per 16x16 (c,i) tile; K = 256 in 8 steps; perm/mask streamed once.
__global__ __launch_bounds__(256) void k_gemm2_update(
        const unsigned short* __restrict__ activeT,
        const unsigned short* __restrict__ xw,
        const float* __restrict__ count,
        const float* __restrict__ perm,
        const unsigned char* __restrict__ mask,
        const int* __restrict__ tp, const int* __restrict__ kp,
        float* __restrict__ d_out) {
    int w  = blockIdx.x * 8 + (threadIdx.x >> 5);
    int l  = threadIdx.x & 31;
    int mt = w >> 9, nt = w & 511;
    int c0 = mt * 16, i0 = nt * 16;
    const unsigned short* arow  = activeT + (size_t)(c0 + (l & 15)) * BATCH;
    const unsigned short* bbase = xw + l * 16;
    int base = (l & 16) >> 1;
    v8f acc = {};
#pragma unroll
    for (int kt = 0; kt < BATCH / 32; ++kt) {
        FragU a, b;
        a.h.lo = *(const u16x8*)(arow + kt * 32 + base);
        a.h.hi = *(const u16x8*)(arow + kt * 32 + base + 16);
        b.u    = *(const u16x16*)(bbase + (size_t)(nt * 8 + kt) * 512);
        acc = __builtin_amdgcn_wmma_f32_16x16x32_bf16(
                  false, a.v, false, b.v, (short)0, acc, false, false);
    }
    float g  = gamma_of(*tp);
    float dm = 2.68e-05f + (0.000134f - 2.68e-05f) * g;
    float dp = 0.015f + dm;
    float* permOut = d_out + 1048576 + (size_t)BATCH * (*kp);
    int n = l & 15, hb = (l & 16) >> 1;
#pragma unroll
    for (int r = 0; r < 8; ++r) {
        int c = c0 + r + hb;
        size_t off = (size_t)c * N_INP + (i0 + n);
        float dP = mask[off] ? (dp * acc[r] - dm * count[c]) : 0.0f;
        permOut[off] = fminf(fmaxf(perm[off] + dP, 0.0f), 1.0f);
    }
}

extern "C" void kernel_launch(void* const* d_in, const int* in_sizes, int n_in,
                              void* d_out, int out_size, void* d_ws, size_t ws_size,
                              hipStream_t stream) {
    const float*         x    = (const float*)d_in[0];
    const float*         perm = (const float*)d_in[1];
    const unsigned char* mask = (const unsigned char*)d_in[2];  // jnp bool: 1B
    const float*         bw   = (const float*)d_in[3];
    const int*           kp   = (const int*)d_in[4];
    const int*           tp   = (const int*)d_in[5];
    float*               out  = (float*)d_out;

    // workspace layout (bytes), ~74 MiB total
    char* ws = (char*)d_ws;
    unsigned short* conn    = (unsigned short*)(ws);                      // 64 MiB
    unsigned short* xT_g1   = (unsigned short*)(ws + 67108864);           //  4 MiB
    unsigned short* xw_g2   = (unsigned short*)(ws + 71303168);           //  4 MiB
    unsigned short* activeT = (unsigned short*)(ws + 75497472);           //  2 MiB
    float*          boost   = (float*)(ws + 77594624);                    // 16 KiB
    float*          count   = (float*)(ws + 77611008);                    // 16 KiB

    k_init<<<(N_COLS * BATCH + 255) / 256, 256, 0, stream>>>(activeT, count);
    k_boost<<<N_COLS, 256, 0, stream>>>(bw, kp, tp, boost);
    k_connected<<<(N_COLS * (N_INP / 8)) / 256, 256, 0, stream>>>(perm, conn);
    k_pack_x_g1<<<(BATCH * N_INP) / 256, 256, 0, stream>>>(x, xT_g1);
    k_pack_x_g2<<<(BATCH * N_INP) / 256, 256, 0, stream>>>(x, xw_g2);
    // 4096 waves: 256 c-tiles x 16 b-tiles
    k_gemm1<<<512, 256, 0, stream>>>(conn, xT_g1, boost, out);
    k_topk<<<BATCH, 256, 0, stream>>>(out, kp, out + 1048576, activeT, count);
    // 131072 waves: 256 c-tiles x 512 i-tiles
    k_gemm2_update<<<16384, 256, 0, stream>>>(activeT, xw_g2, count, perm,
                                              mask, tp, kp, out);
}
